// WaveConv2d_59055800320802
// MI455X (gfx1250) — matrix operand order repaired and checked by reading, not compiled, not run on hardware
//
#include <hip/hip_runtime.h>

// ---------------------------------------------------------------------------
// WaveConv2d for MI455X (gfx1250, wave32).
// 4-level db6 DWT -> per-pixel 32x32 channel mix (v_wmma_f32_16x16x32_f16)
// -> 4-level inverse DWT.
// Each DWT/IDWT level is a single LDS-fused row+column kernel: no row
// intermediates ever touch HBM. Total traffic ~1.5 GB => ~65us at 23.3 TB/s.
// ---------------------------------------------------------------------------

typedef __attribute__((ext_vector_type(16))) _Float16 v16h;
typedef __attribute__((ext_vector_type(8)))  float    v8f;

// db6 filters (pytorch_wavelets semantics; XLA conv = cross-correlation).
__constant__ float ANA0[12] = {
  0.11154074335008017f, 0.4946238903983854f, 0.7511339080215775f, 0.3152503517092432f,
 -0.22626469396516913f,-0.12976686756709563f, 0.09750160558707936f, 0.02752286553001629f,
 -0.031582039318031156f, 0.0005538422009938016f, 0.004777257511010651f,-0.00107730108499558f };
__constant__ float ANA1[12] = {
 -0.00107730108499558f,-0.004777257511010651f, 0.0005538422009938016f, 0.031582039318031156f,
  0.02752286553001629f,-0.09750160558707936f,-0.12976686756709563f, 0.22626469396516913f,
  0.3152503517092432f,-0.7511339080215775f, 0.4946238903983854f,-0.11154074335008017f };
__constant__ float SYN0[12] = {
 -0.00107730108499558f, 0.004777257511010651f, 0.0005538422009938016f,-0.031582039318031156f,
  0.02752286553001629f, 0.09750160558707936f,-0.12976686756709563f,-0.22626469396516913f,
  0.3152503517092432f, 0.7511339080215775f, 0.4946238903983854f, 0.11154074335008017f };
__constant__ float SYN1[12] = {
 -0.11154074335008017f, 0.4946238903983854f,-0.7511339080215775f, 0.3152503517092432f,
  0.22626469396516913f,-0.12976686756709563f,-0.09750160558707936f, 0.02752286553001629f,
  0.031582039318031156f, 0.0005538422009938016f,-0.004777257511010651f,-0.00107730108499558f };

__device__ __forceinline__ int symref(int u, int N) {
  // 'symmetric' (edge-repeating) reflection: ...cba|abc...|cba...
  while (u < 0 || u >= N) {
    if (u < 0)  u = -u - 1;
    if (u >= N) u = 2 * N - 1 - u;
  }
  return u;
}

// ---------------------------------------------------------------------------
// Fused analysis level: in (NIMG,N,N) -> ll,lh,hl,hh (NIMG,No,No), padL = 10.
// One block = 16x16 output tile of one image. Input halo tile = 42x42.
// Row filter commutes with row reflection, so reflecting raw input rows at
// load time yields exactly the reflected lo/hi rows the column filter needs.
// ---------------------------------------------------------------------------
__global__ __launch_bounds__(256)
void afb2(const float* __restrict__ in, int N,
          float* __restrict__ ll, float* __restrict__ lh,
          float* __restrict__ hl, float* __restrict__ hh, int No) {
  __shared__ float sIn[42][44];   // padded stride vs 64 banks
  __shared__ float sLo[42][16];
  __shared__ float sHi[42][16];

  const int img  = blockIdx.z;
  const int j0   = blockIdx.x * 16;       // output col base
  const int i0   = blockIdx.y * 16;       // output row base
  const int xin0 = 2 * j0 - 10;
  const int yin0 = 2 * i0 - 10;
  const float* src = in + (size_t)img * N * N;

  // load 42x42 halo tile with symmetric reflection
  for (int e = threadIdx.x; e < 42 * 42; e += 256) {
    int r = e / 42, c = e - r * 42;
    sIn[r][c] = src[(size_t)symref(yin0 + r, N) * N + symref(xin0 + c, N)];
  }
  __syncthreads();

  // row filter: 42 rows x 16 output cols -> lo/hi
  for (int e = threadIdx.x; e < 42 * 16; e += 256) {
    int r = e >> 4, j = e & 15;
    float a = 0.f, b = 0.f;
#pragma unroll
    for (int t = 0; t < 12; ++t) {
      float v = sIn[r][2 * j + t];        // local col = 2*j + t
      a += v * ANA0[t];
      b += v * ANA1[t];
    }
    sLo[r][j] = a;
    sHi[r][j] = b;
  }
  __syncthreads();

  // column filter: one output pixel per thread, all 4 subbands
  int il = threadIdx.x >> 4, jl = threadIdx.x & 15;
  float vll = 0.f, vlh = 0.f, vhl = 0.f, vhh = 0.f;
#pragma unroll
  for (int t = 0; t < 12; ++t) {
    float lo = sLo[2 * il + t][jl];       // local row = 2*il + t
    float hi = sHi[2 * il + t][jl];
    vll += lo * ANA0[t]; vlh += lo * ANA1[t];
    vhl += hi * ANA0[t]; vhh += hi * ANA1[t];
  }
  int oi = i0 + il, oj = j0 + jl;
  if (oi < No && oj < No) {
    size_t o = (size_t)img * No * No + (size_t)oi * No + oj;
    ll[o] = vll; lh[o] = vlh; hl[o] = vhl; hh[o] = vhh;
  }
}

// ---------------------------------------------------------------------------
// Fused synthesis level: ll(strided, possibly cropped) + lh,hl,hh (NIMG,Nin,Nin)
// -> out (NIMG,Nout,Nout), Nout = 2*Nin-10.
// One block = 32x32 output tile. Needs a 22x22 tile of each input band
// (rows rb..rb+21, cols ib..ib+21 with rb=u0/2-1, ib=v0/2-1; zero outside
// [0,Nin) = transposed-conv boundary). Stage 1: column synthesis -> 32x22
// lo/hi. Stage 2: row synthesis -> 32x32 outputs.
// ---------------------------------------------------------------------------
__global__ __launch_bounds__(256)
void sfb2(const float* __restrict__ ll, int llImgS, int llRowS,
          const float* __restrict__ lh, const float* __restrict__ hl,
          const float* __restrict__ hh, int Nin,
          float* __restrict__ out, int Nout) {
  __shared__ float sLL[22][24], sLH[22][24], sHL[22][24], sHH[22][24];
  __shared__ float sLo[32][24], sHi[32][24];

  const int img = blockIdx.z;
  const int v0  = blockIdx.x * 32;        // output col base
  const int u0  = blockIdx.y * 32;        // output row base
  const int ib  = v0 / 2 - 1;             // first needed input col (may be -1)
  const int rb  = u0 / 2 - 1;             // first needed input row (may be -1)
  const size_t bImgS = (size_t)Nin * Nin;

  for (int e = threadIdx.x; e < 22 * 22; e += 256) {
    int r = e / 22, i = e - r * 22;
    int gr = rb + r, gi = ib + i;
    bool ok = (gr >= 0) && (gr < Nin) && (gi >= 0) && (gi < Nin);
    float vl = 0.f, v1 = 0.f, v2 = 0.f, v3 = 0.f;
    if (ok) {
      size_t bo = (size_t)img * bImgS + (size_t)gr * Nin + gi;
      vl = ll[(size_t)img * llImgS + (size_t)gr * llRowS + gi];
      v1 = lh[bo]; v2 = hl[bo]; v3 = hh[bo];
    }
    sLL[r][i] = vl; sLH[r][i] = v1; sHL[r][i] = v2; sHH[r][i] = v3;
  }
  __syncthreads();

  // stage 1: column synthesis, lo/hi at 32 output rows x 22 input cols
  for (int e = threadIdx.x; e < 32 * 22; e += 256) {
    int ul = e / 22, il = e - ul * 22;
    int u = u0 + ul;
    float aL = 0.f, aH = 0.f;
#pragma unroll
    for (int t = 0; t < 12; ++t) {
      int s = u + t - 1;                  // pos in (pad1 + 2x-dilated) input
      if (s >= 0 && !(s & 1)) {
        int r = (s >> 1) - rb;            // local row, always in [0,21]
        aL += sLL[r][il] * SYN0[t] + sLH[r][il] * SYN1[t];
        aH += sHL[r][il] * SYN0[t] + sHH[r][il] * SYN1[t];
      }
    }
    sLo[ul][il] = aL;
    sHi[ul][il] = aH;
  }
  __syncthreads();

  // stage 2: row synthesis -> 32x32 outputs
  for (int e = threadIdx.x; e < 32 * 32; e += 256) {
    int ul = e >> 5, vl = e & 31;
    int v = v0 + vl;
    float acc = 0.f;
#pragma unroll
    for (int t = 0; t < 12; ++t) {
      int s = v + t - 1;
      if (s >= 0 && !(s & 1)) {
        int i = (s >> 1) - ib;            // local col, always in [0,21]
        acc += sLo[ul][i] * SYN0[t] + sHi[ul][i] * SYN1[t];
      }
    }
    int ou = u0 + ul, ov = v0 + vl;
    if (ou < Nout && ov < Nout)
      out[(size_t)img * Nout * Nout + (size_t)ou * Nout + ov] = acc;
  }
}

// ---------------------------------------------------------------------------
// Per-pixel channel mix via WMMA: out[b,o] = sum_i X[b,i] * W[i,o] at each of
// 42x42 pixels, 4 subbands. One wave per (pixel, subband, 16-wide N tile).
// M=16 (B=8 valid rows, rest zero), K=32=CIN in one v_wmma_f32_16x16x32_f16.
// ---------------------------------------------------------------------------
__global__ __launch_bounds__(256)
void mix_wmma(const float* __restrict__ yl, const float* __restrict__ lh,
              const float* __restrict__ hl, const float* __restrict__ hh,
              const float* __restrict__ wyl, const float* __restrict__ wyh,
              float* __restrict__ oyl, float* __restrict__ olh,
              float* __restrict__ ohl, float* __restrict__ ohh) {
  const int P = 42 * 42;
  int pix  = blockIdx.x;                  // 0..1763
  int lane = threadIdx.x & 31;
  int wid  = threadIdx.x >> 5;            // 8 waves per block
  int s    = wid >> 1;                    // subband: 0=yl, 1=lh, 2=hl, 3=hh
  int nt   = wid & 1;                     // N tile (cols 0-15 / 16-31)

  const float* X; float* O;
  if      (s == 0) { X = yl; O = oyl; }
  else if (s == 1) { X = lh; O = olh; }
  else if (s == 2) { X = hl; O = ohl; }
  else             { X = hh; O = ohh; }

  int m  = lane & 15;                     // A row (batch)
  int hf = lane >> 4;                     // lane half

  // A (16x32 f16): VGPR v holds K=2v,2v+1 (+16 for v>=4); upper lanes +8.
  v16h a;
#pragma unroll
  for (int e = 0; e < 16; ++e) {
    int vg = e >> 1;
    int k  = ((vg >= 4) ? 16 : 0) + (hf ? 8 : 0) + ((vg & 3) << 1) + (e & 1);
    float v = (m < 8) ? X[(size_t)(m * 32 + k) * P + pix] : 0.f;
    a[e] = (_Float16)v;
  }

  // B (32x16 f16): N = lane%16, same K order as A: K=0..7,16..23 (lanes 0-15), K=8..15,24..31 (lanes 16-31).
  int o  = nt * 16 + (lane & 15);
  int kb = hf ? 16 : 0;
  v16h b;
#pragma unroll
  for (int e = 0; e < 16; ++e) {
    int k = ((e >> 3) << 4) + (kb >> 1) + (e & 7);
    float wv = (s == 0) ? wyl[(size_t)(k * 32 + o) * P + pix]
                        : wyh[((size_t)(k * 32 + o) * 3 + (s - 1)) * P + pix];
    b[e] = (_Float16)wv;
  }

  v8f c = {};
  c = __builtin_amdgcn_wmma_f32_16x16x32_f16(false, a, false, b, (short)0, c,
                                             false, false);

  // D: VGPR r -> row r (lanes 0-15) / r+8 (lanes 16-31); rows 0..7 valid.
  if (!hf) {
    int n = lane;
#pragma unroll
    for (int r = 0; r < 8; ++r)
      O[(size_t)(r * 32 + nt * 16 + n) * P + pix] = c[r];
  }
}

// ---------------------------------------------------------------------------
extern "C" void kernel_launch(void* const* d_in, const int* in_sizes, int n_in,
                              void* d_out, int out_size, void* d_ws, size_t ws_size,
                              hipStream_t stream) {
  const float* x   = (const float*)d_in[0];   // (8,32,512,512)
  const float* wyl = (const float*)d_in[1];   // (32,32,42,42)
  const float* wyh = (const float*)d_in[2];   // (32,32,3,42,42)
  float* out = (float*)d_out;                 // (8,32,512,512)
  float* ws  = (float*)d_ws;

  const int NI = 256;                         // B*CIN image planes
  // level sizes: 512 -> 261 -> 136 -> 73 -> 42 ; padL = 10 at every level
  const int N0 = 512, N1 = 261, N2 = 136, N3 = 73, N4 = 42;

  // ---- workspace layout (floats), ~386 MB total ----
  size_t off = 0;
  auto A = [&](size_t n) { size_t r = off; off += n; return r; };
  const size_t LLa = A((size_t)NI * 262 * 262);   // ll ping (261^2, 74^2, 262^2)
  const size_t LLb = A((size_t)NI * N2 * N2);     // ll pong (136^2)
  const size_t LH1 = A((size_t)NI * N1 * N1), HL1 = A((size_t)NI * N1 * N1), HH1 = A((size_t)NI * N1 * N1);
  const size_t LH2 = A((size_t)NI * N2 * N2), HL2 = A((size_t)NI * N2 * N2), HH2 = A((size_t)NI * N2 * N2);
  const size_t LH3 = A((size_t)NI * N3 * N3), HL3 = A((size_t)NI * N3 * N3), HH3 = A((size_t)NI * N3 * N3);
  const size_t YL  = A((size_t)NI * N4 * N4), LH4 = A((size_t)NI * N4 * N4);
  const size_t HL4 = A((size_t)NI * N4 * N4), HH4 = A((size_t)NI * N4 * N4);
  const size_t YLM = A((size_t)NI * N4 * N4), LH4M = A((size_t)NI * N4 * N4);
  const size_t HL4M= A((size_t)NI * N4 * N4), HH4M = A((size_t)NI * N4 * N4);
  (void)ws_size; (void)in_sizes; (void)n_in; (void)out_size;

  auto tg = [](int n, int T) { return (unsigned)((n + T - 1) / T); };

  // ================= forward DWT (fused row+col per level) =================
  afb2<<<dim3(tg(N1,16), tg(N1,16), NI), 256, 0, stream>>>(
      x, N0, ws + LLa, ws + LH1, ws + HL1, ws + HH1, N1);
  afb2<<<dim3(tg(N2,16), tg(N2,16), NI), 256, 0, stream>>>(
      ws + LLa, N1, ws + LLb, ws + LH2, ws + HL2, ws + HH2, N2);
  afb2<<<dim3(tg(N3,16), tg(N3,16), NI), 256, 0, stream>>>(
      ws + LLb, N2, ws + LLa, ws + LH3, ws + HL3, ws + HH3, N3);
  afb2<<<dim3(tg(N4,16), tg(N4,16), NI), 256, 0, stream>>>(
      ws + LLa, N3, ws + YL, ws + LH4, ws + HL4, ws + HH4, N4);

  // ================= WMMA channel mix (coarsest subbands) =================
  mix_wmma<<<dim3(N4 * N4), 256, 0, stream>>>(
      ws + YL, ws + LH4, ws + HL4, ws + HH4, wyl, wyh,
      ws + YLM, ws + LH4M, ws + HL4M, ws + HH4M);

  // ================= inverse DWT (fused col+row per level) =================
  // level 4: 42 -> 74 (into LLa, stored 74x74)
  sfb2<<<dim3(tg(74,32), tg(74,32), NI), 256, 0, stream>>>(
      ws + YLM, N4 * N4, N4, ws + LH4M, ws + HL4M, ws + HH4M, N4, ws + LLa, 74);
  // level 3: ll cropped 74->73, 73 -> 136 (into LLb)
  sfb2<<<dim3(tg(N2,32), tg(N2,32), NI), 256, 0, stream>>>(
      ws + LLa, 74 * 74, 74, ws + LH3, ws + HL3, ws + HH3, N3, ws + LLb, N2);
  // level 2: 136 -> 262 (into LLa, stored 262x262)
  sfb2<<<dim3(tg(262,32), tg(262,32), NI), 256, 0, stream>>>(
      ws + LLb, N2 * N2, N2, ws + LH2, ws + HL2, ws + HH2, N2, ws + LLa, 262);
  // level 1: ll cropped 262->261, 261 -> 512 (straight into d_out)
  sfb2<<<dim3(tg(N0,32), tg(N0,32), NI), 256, 0, stream>>>(
      ws + LLa, 262 * 262, 262, ws + LH1, ws + HL1, ws + HH1, N1, out, N0);
}
